// VRWKV_SpatialMix_V6_25331717112517
// MI455X (gfx1250) — matrix operand order repaired and checked by reading, not compiled, not run on hardware
//
#include <hip/hip_runtime.h>
#include <hip/hip_bf16.h>
#include <math.h>

#define Bb 8
#define Tt 2048
#define Cc 512
#define Hh 16
#define HS 32
#define TM 32
#define TD 64
#define Mrows (Bb*Tt)   // 16384
#define WCHUNK 8        // timesteps staged in LDS per wkv iteration

typedef __attribute__((ext_vector_type(16))) __bf16 v16bf;
typedef __attribute__((ext_vector_type(8)))  __bf16 v8bf;
typedef __attribute__((ext_vector_type(8)))  float  v8f;
typedef unsigned short u16;

#if defined(__gfx1250__) && __has_builtin(__builtin_amdgcn_global_load_async_to_lds_b32)
#define HAVE_ASYNC_LDS 1
#else
#define HAVE_ASYNC_LDS 0
#endif

__device__ __forceinline__ u16 f2bf(float f) {
  __bf16 b = (__bf16)f;
  return __builtin_bit_cast(u16, b);
}
__device__ __forceinline__ float bf2f(u16 u) {
  __bf16 b = __builtin_bit_cast(__bf16, u);
  return (float)b;
}

// ---------------- fp32 -> bf16 weight conversion ----------------
__global__ void cvt_bf16_kernel(const float* __restrict__ src, u16* __restrict__ dst, int n) {
  int i = blockIdx.x * blockDim.x + threadIdx.x;
  if (i < n) dst[i] = f2bf(src[i]);
}

// ---------------- token-shift + 5-way data-dependent mixing ----------------
__global__ void mix_kernel(const float* __restrict__ x,
                           const float* __restrict__ maa_x,
                           const float* __restrict__ maa_w,
                           const float* __restrict__ maa_k,
                           const float* __restrict__ maa_v,
                           const float* __restrict__ maa_r,
                           const float* __restrict__ maa_g,
                           const float* __restrict__ w1,   // [C, 160]
                           const float* __restrict__ w2,   // [5, 32, C]
                           u16* __restrict__ oxw,
                           u16* __restrict__ oxk,
                           u16* __restrict__ oxv,
                           u16* __restrict__ oxr,
                           u16* __restrict__ oxg) {
  __shared__ float sx[Cc], sxx[Cc], sxxx[Cc], sh[5*TM];
  int row = blockIdx.x;
  int t = row % Tt;
  const float* xrow  = x + (size_t)row * Cc;
  const float* xprev = xrow - Cc;  // only read when t > 0
  for (int c = threadIdx.x; c < Cc; c += blockDim.x) {
    float xc = xrow[c];
    float xp = (t == 0) ? 0.f : xprev[c];
    float xxc = xp - xc;
    sx[c] = xc; sxx[c] = xxc; sxxx[c] = xc + xxc * maa_x[c];
  }
  __syncthreads();
  for (int j = threadIdx.x; j < 5*TM; j += blockDim.x) {
    float acc = 0.f;
    for (int c = 0; c < Cc; ++c) acc += sxxx[c] * w1[c * (5*TM) + j];
    sh[j] = tanhf(acc);
  }
  __syncthreads();
  size_t base = (size_t)row * Cc;
  for (int c = threadIdx.x; c < Cc; c += blockDim.x) {
    float xc = sx[c], xxc = sxx[c];
    float m[5];
#pragma unroll
    for (int f = 0; f < 5; ++f) {
      float acc = 0.f;
#pragma unroll
      for (int d = 0; d < TM; ++d) acc += sh[f*TM + d] * w2[(size_t)(f*TM + d) * Cc + c];
      m[f] = acc;
    }
    oxw[base + c] = f2bf(xc + xxc * (maa_w[c] + m[0]));
    oxk[base + c] = f2bf(xc + xxc * (maa_k[c] + m[1]));
    oxv[base + c] = f2bf(xc + xxc * (maa_v[c] + m[2]));
    oxr[base + c] = f2bf(xc + xxc * (maa_r[c] + m[3]));
    oxg[base + c] = f2bf(xc + xxc * (maa_g[c] + m[4]));
  }
}

// ---------------- decay path: ds = exp(-exp(td + tanh(xw @ w1) @ w2)) ----------------
__global__ void decay_kernel(const u16* __restrict__ xw,
                             const float* __restrict__ td,    // [C]
                             const float* __restrict__ tw1,   // [C, 64]
                             const float* __restrict__ tw2,   // [64, C]
                             float* __restrict__ dsb) {
  __shared__ float sxw[Cc], shd[TD];
  int row = blockIdx.x;
  size_t base = (size_t)row * Cc;
  for (int c = threadIdx.x; c < Cc; c += blockDim.x) sxw[c] = bf2f(xw[base + c]);
  __syncthreads();
  for (int d = threadIdx.x; d < TD; d += blockDim.x) {
    float acc = 0.f;
    for (int c = 0; c < Cc; ++c) acc += sxw[c] * tw1[c * TD + d];
    shd[d] = tanhf(acc);
  }
  __syncthreads();
  for (int c = threadIdx.x; c < Cc; c += blockDim.x) {
    float acc = td[c];
#pragma unroll 8
    for (int d = 0; d < TD; ++d) acc += shd[d] * tw2[d * Cc + c];
    dsb[base + c] = expf(-expf(acc));
  }
}

// ---------------- bf16 WMMA GEMM: Out[M,512] = A[M,512] @ W[512,512]^T ----------------
// One wave -> 16x32 output strip (2 accumulators share one A fragment).
// 8 waves/block -> 16x256 strip; grid = (M/16, 2).
__global__ void gemm_bf16_wmma(const u16* __restrict__ A,
                               const u16* __restrict__ W,
                               float* __restrict__ Out,
                               int apply_silu) {
  int wav  = threadIdx.x >> 5;
  int lane = threadIdx.x & 31;
  int hlf  = lane >> 4;
  int lq   = lane & 15;
  int m0 = blockIdx.x * 16;
  int n0 = (blockIdx.y * 16 + wav * 2) * 16;   // two adjacent 16-wide tiles
  const u16* arow  = A + (size_t)(m0 + lq) * Cc;
  const u16* brow0 = W + (size_t)(n0 + lq) * Cc;
  const u16* brow1 = brow0 + (size_t)16 * Cc;
  v8f acc0 = {}, acc1 = {};
  for (int k0 = 0; k0 < Cc; k0 += 32) {
    // A: 16x32 bf16; lane half 0 holds K {0..15}, half 1 holds K {16..31}
    union { v16bf v; v8bf h[2]; } a;
    a.h[0] = *(const v8bf*)(arow + k0 + hlf * 16);
    a.h[1] = *(const v8bf*)(arow + k0 + hlf * 16 + 8);
    // B: 32x16 bf16 (column n = row of W); lane half selects 16-wide K block
    v16bf bv0 = *(const v16bf*)(brow0 + k0 + hlf * 16);
    v16bf bv1 = *(const v16bf*)(brow1 + k0 + hlf * 16);
    if (k0 + 64 < Cc) {
      __builtin_prefetch(arow  + k0 + 64, 0, 3);
      __builtin_prefetch(brow0 + k0 + 64, 0, 3);
      __builtin_prefetch(brow1 + k0 + 64, 0, 3);
    }
    acc0 = __builtin_amdgcn_wmma_f32_16x16x32_bf16(false, a.v, false, bv0,
                                                   (short)0, acc0, false, false);
    acc1 = __builtin_amdgcn_wmma_f32_16x16x32_bf16(false, a.v, false, bv1,
                                                   (short)0, acc1, false, false);
  }
#pragma unroll
  for (int i = 0; i < 8; ++i) {
    size_t rbase = (size_t)(m0 + hlf * 8 + i) * Cc;
    float o0 = acc0[i];
    float o1 = acc1[i];
    if (apply_silu) {
      o0 = o0 / (1.f + expf(-o0));
      o1 = o1 / (1.f + expf(-o1));
    }
    Out[rbase + n0 + lq]      = o0;
    Out[rbase + n0 + 16 + lq] = o1;
  }
}

// ---------------- WKV6 recurrence ----------------
// One 1024-thread block per (b,h): wave index = v-dim i, lane = k-dim j,
// one state element S[j][i] per thread. Per step: ~8 VALU + 5-step wave
// reduction. WCHUNK timesteps of r/k/v/decay staged into LDS via the gfx1250
// async global->LDS path (ASYNCcnt) when available.
__global__ __launch_bounds__(1024) void wkv_kernel(const float* __restrict__ r,
                                                   const float* __restrict__ k,
                                                   const float* __restrict__ v,
                                                   const float* __restrict__ dsb,
                                                   const float* __restrict__ u,  // [H, HS]
                                                   float* __restrict__ y) {
  __shared__ float stage[4 * WCHUNK * HS];  // r | k | ds | v, each [WCHUNK][HS]
  float* sr = stage;
  float* sk = stage + WCHUNK * HS;
  float* sd = stage + 2 * WCHUNK * HS;
  float* sv = stage + 3 * WCHUNK * HS;

  int bh = blockIdx.x;
  int b = bh / Hh, h = bh % Hh;
  int i = threadIdx.x >> 5;   // wave id  -> v index
  int j = threadIdx.x & 31;   // lane     -> k index
  float uj = u[h * HS + j];
  float S = 0.f;

  // staging assignment: 1024 threads <-> 4 arrays x WCHUNK x 32 elements
  int tid = threadIdx.x;
  int arr = tid >> 8;         // 0..3
  int e   = tid & 255;        // 0..WCHUNK*32-1
  int stt = e >> 5, sc = e & 31;
  const float* sgsrc = (arr == 0) ? r : (arr == 1) ? k : (arr == 2) ? dsb : v;
  float* sgdst = stage + arr * (WCHUNK * HS) + e;

  size_t base = ((size_t)b * Tt) * Cc + h * HS;
  for (int t0 = 0; t0 < Tt; t0 += WCHUNK) {
    const float* gsrc = sgsrc + base + (size_t)stt * Cc + sc;
#if HAVE_ASYNC_LDS
    __builtin_amdgcn_global_load_async_to_lds_b32(
        (__attribute__((address_space(1))) int*)(unsigned long long)(const void*)gsrc,
        (__attribute__((address_space(3))) int*)(unsigned int)(unsigned long long)(void*)sgdst,
        0, 0);
#if __has_builtin(__builtin_amdgcn_s_wait_asynccnt)
    __builtin_amdgcn_s_wait_asynccnt(0);
#else
    asm volatile("s_wait_asynccnt 0" ::: "memory");
#endif
#else
    *sgdst = *gsrc;
#endif
    __syncthreads();
#pragma unroll
    for (int tt = 0; tt < WCHUNK; ++tt) {
      float rj = sr[tt * HS + j];
      float kj = sk[tt * HS + j];
      float dj = sd[tt * HS + j];
      float vi = sv[tt * HS + i];
      float kv = kj * vi;
      float contrib = rj * (S + uj * kv);
#pragma unroll
      for (int off = 16; off > 0; off >>= 1) contrib += __shfl_xor(contrib, off, 32);
      if (j == 0) y[base + (size_t)tt * Cc + i] = contrib;
      S = dj * S + kv;
    }
    __syncthreads();
    base += (size_t)WCHUNK * Cc;
  }
}

// ---------------- per-head GroupNorm + gate; z = yn * silu(g) -> bf16 ----------------
__global__ void gnorm_gate_kernel(const float* __restrict__ y,
                                  const float* __restrict__ g,  // silu already applied
                                  const float* __restrict__ lnw,
                                  const float* __restrict__ lnb,
                                  u16* __restrict__ zb) {
  int row  = blockIdx.x;
  int wav  = threadIdx.x >> 5;
  int lane = threadIdx.x & 31;
  int c = wav * HS + lane;
  size_t idx = (size_t)row * Cc + c;
  float yv = y[idx];
  float s = yv, s2 = yv * yv;
#pragma unroll
  for (int off = 16; off > 0; off >>= 1) {
    s  += __shfl_xor(s,  off, 32);
    s2 += __shfl_xor(s2, off, 32);
  }
  float mu  = s  * (1.f / HS);
  float var = s2 * (1.f / HS) - mu * mu;
  float yn = (yv - mu) * rsqrtf(var + 1e-5f) * lnw[c] + lnb[c];
  zb[idx] = f2bf(yn * g[idx]);
}

extern "C" void kernel_launch(void* const* d_in, const int* in_sizes, int n_in,
                              void* d_out, int out_size, void* d_ws, size_t ws_size,
                              hipStream_t stream) {
  (void)in_sizes; (void)n_in; (void)out_size; (void)ws_size;
  const float* x     = (const float*)d_in[0];
  const float* maa_x = (const float*)d_in[1];
  const float* maa_w = (const float*)d_in[2];
  const float* maa_k = (const float*)d_in[3];
  const float* maa_v = (const float*)d_in[4];
  const float* maa_r = (const float*)d_in[5];
  const float* maa_g = (const float*)d_in[6];
  const float* w1    = (const float*)d_in[7];
  const float* w2    = (const float*)d_in[8];
  const float* tdec  = (const float*)d_in[9];
  const float* tdw1  = (const float*)d_in[10];
  const float* tdw2  = (const float*)d_in[11];
  const float* faaaa = (const float*)d_in[12];
  const float* Wsrc[5] = {(const float*)d_in[13], (const float*)d_in[14],
                          (const float*)d_in[15], (const float*)d_in[16],
                          (const float*)d_in[17]};  // Wr, Wk, Wv, Wg, Wo
  const float* lnw = (const float*)d_in[18];
  const float* lnb = (const float*)d_in[19];

  // ---- carve workspace ----
  char* p = (char*)d_ws;
  auto carve = [&](size_t bytes) -> char* {
    char* r = p;
    p += (bytes + 255) & ~(size_t)255;
    return r;
  };
  const size_t MC = (size_t)Mrows * Cc;
  u16* Wb[5];
  for (int i = 0; i < 5; ++i) Wb[i] = (u16*)carve((size_t)Cc * Cc * sizeof(u16));
  u16* xwb = (u16*)carve(MC * sizeof(u16));
  u16* xkb = (u16*)carve(MC * sizeof(u16));
  u16* xvb = (u16*)carve(MC * sizeof(u16));
  u16* xrb = (u16*)carve(MC * sizeof(u16));
  u16* xgb = (u16*)carve(MC * sizeof(u16));
  float* dsb = (float*)carve(MC * sizeof(float));
  float* rb  = (float*)carve(MC * sizeof(float));
  float* kb  = (float*)carve(MC * sizeof(float));
  float* vb  = (float*)carve(MC * sizeof(float));
  float* gb  = (float*)carve(MC * sizeof(float));
  float* yb  = (float*)carve(MC * sizeof(float));
  u16*   zbb = (u16*)carve(MC * sizeof(u16));

  // ---- 1) weights fp32 -> bf16 ----
  for (int i = 0; i < 5; ++i)
    cvt_bf16_kernel<<<(Cc * Cc + 255) / 256, 256, 0, stream>>>(Wsrc[i], Wb[i], Cc * Cc);

  // ---- 2) token shift + mixing -> bf16 activations ----
  mix_kernel<<<Mrows, 256, 0, stream>>>(x, maa_x, maa_w, maa_k, maa_v, maa_r, maa_g,
                                        w1, w2, xwb, xkb, xvb, xrb, xgb);

  // ---- 3) decay ds = exp(-exp(w)) ----
  decay_kernel<<<Mrows, 64, 0, stream>>>(xwb, tdec, tdw1, tdw2, dsb);

  // ---- 4) projections via WMMA (r, k, v, g=silu) ----
  dim3 ggrid(Mrows / 16, Cc / 256);  // (1024, 2), 8 waves/block, 2 tiles/wave
  gemm_bf16_wmma<<<ggrid, 256, 0, stream>>>(xrb, Wb[0], rb, 0);
  gemm_bf16_wmma<<<ggrid, 256, 0, stream>>>(xkb, Wb[1], kb, 0);
  gemm_bf16_wmma<<<ggrid, 256, 0, stream>>>(xvb, Wb[2], vb, 0);
  gemm_bf16_wmma<<<ggrid, 256, 0, stream>>>(xgb, Wb[3], gb, 1);

  // ---- 5) WKV6 recurrence (1024 threads per (b,h)) ----
  wkv_kernel<<<Bb * Hh, 1024, 0, stream>>>(rb, kb, vb, dsb, faaaa, yb);

  // ---- 6) groupnorm + gate -> bf16 ----
  gnorm_gate_kernel<<<Mrows, Cc, 0, stream>>>(yb, gb, lnw, lnb, zbb);

  // ---- 7) output projection via WMMA -> d_out (fp32) ----
  gemm_bf16_wmma<<<ggrid, 256, 0, stream>>>(zbb, Wb[4], (float*)d_out, 0);
}
